// SelectK_38852274160186
// MI455X (gfx1250) — matrix operand order repaired and checked by reading, not compiled, not run on hardware
//
#include <hip/hip_runtime.h>
#include <hip/hip_fp16.h>
#include <math.h>

// Problem constants (match reference setup_inputs)
#define NROWS   4096
#define HHDIM   256
#define SSEN    30000
#define VVOC    30000
#define NNBR    32
#define KTOP    8
#define WPR     940          // u32 words per row bitmask (ceil(30000/32)=938, padded)
#define FEPS    1e-8f
#define LOGEPS  (-18.420680743952367f)   // log(1e-8)
#define NTILES  3            // output 16x16 tiles per wave along N (1875 = 3*625)

typedef __attribute__((ext_vector_type(16))) _Float16 v16h;
typedef __attribute__((ext_vector_type(8)))  _Float16 v8h;
typedef __attribute__((ext_vector_type(8)))  float    v8f;

// ---------------------------------------------------------------- conversions
__global__ void cvt_f32_f16_kernel(const float* __restrict__ src,
                                   _Float16* __restrict__ dst, int n) {
    int i = blockIdx.x * blockDim.x + threadIdx.x;
    if (i < n) dst[i] = (_Float16)src[i];
}

// W[256,30000] f32 (row-major, K outer) -> Wt[30000,256] f16 (N outer, K inner)
// 16x16 LDS-tiled transpose: grid (30000/16, 256/16), block (16,16)
__global__ void transpose_cvt_kernel(const float* __restrict__ W,
                                     _Float16* __restrict__ Wt) {
    __shared__ _Float16 tile[16][17];
    const int nbase = blockIdx.x * 16;
    const int kbase = blockIdx.y * 16;
    const int tx = threadIdx.x, ty = threadIdx.y;
    tile[ty][tx] = (_Float16)W[(size_t)(kbase + ty) * SSEN + (nbase + tx)];
    __syncthreads();
    Wt[(size_t)(nbase + ty) * HHDIM + (kbase + tx)] = tile[tx][ty];
}

__global__ void copy_f32x4_kernel(const float4* __restrict__ src,
                                  float4* __restrict__ dst, size_t n4) {
    size_t i = (size_t)blockIdx.x * blockDim.x + threadIdx.x;
    if (i < n4) dst[i] = src[i];
}

// ---------------------------------------------------------------- top-8 per row
__global__ void topk_kernel(const float* __restrict__ preds,
                            int* __restrict__ topk) {
    const int row = blockIdx.x;
    const int tid = threadIdx.x;
    const float* p = preds + (size_t)row * VVOC;

    float lv[KTOP];
    int   li[KTOP];
#pragma unroll
    for (int i = 0; i < KTOP; ++i) { lv[i] = -INFINITY; li[i] = 0x7fffffff; }

    for (int j = tid; j < VVOC; j += 256) {
        float v = p[j];
        if (v > lv[KTOP - 1] || (v == lv[KTOP - 1] && j < li[KTOP - 1])) {
            lv[KTOP - 1] = v; li[KTOP - 1] = j;
            for (int q = KTOP - 1; q > 0; --q) {
                bool sw = (lv[q] > lv[q - 1]) ||
                          (lv[q] == lv[q - 1] && li[q] < li[q - 1]);
                if (!sw) break;
                float tv = lv[q]; lv[q] = lv[q - 1]; lv[q - 1] = tv;
                int   ti = li[q]; li[q] = li[q - 1]; li[q - 1] = ti;
            }
        }
    }

    __shared__ float sv[256];
    __shared__ int   si[256];
    int ptr = 0;
    for (int r = 0; r < KTOP; ++r) {
        sv[tid] = (ptr < KTOP) ? lv[ptr] : -INFINITY;
        si[tid] = (ptr < KTOP) ? li[ptr] : 0x7fffffff;
        __syncthreads();
        for (int off = 128; off > 0; off >>= 1) {
            if (tid < off) {
                float v2 = sv[tid + off]; int i2 = si[tid + off];
                if (v2 > sv[tid] || (v2 == sv[tid] && i2 < si[tid])) {
                    sv[tid] = v2; si[tid] = i2;
                }
            }
            __syncthreads();
        }
        int widx = si[0];
        __syncthreads();                 // everyone read winner before reuse
        if (ptr < KTOP && li[ptr] == widx) ++ptr;   // owner advances
        if (tid == 0) topk[row * KTOP + r] = widx;
    }
}

// ---------------------------------------------------------------- mask build
__global__ void mask_kernel(const int* __restrict__ grapharea,
                            const int* __restrict__ topk,
                            unsigned int* __restrict__ maskbits,
                            int* __restrict__ nsel) {
    const int row = blockIdx.x;
    const int tid = threadIdx.x;
    __shared__ unsigned int m[WPR];
    __shared__ int cnt[256];

    for (int w = tid; w < WPR; w += 256) m[w] = 0u;
    __syncthreads();

    // 8 top indices x 32 neighbours = 256 entries, one per thread
    const int k  = tid >> 5;
    const int nb = tid & 31;
    const int gidx = topk[row * KTOP + k];
    const int g = grapharea[(size_t)(SSEN + gidx) * NNBR + nb] - 1;
    if (g >= 0 && g < SSEN) atomicOr(&m[g >> 5], 1u << (g & 31));
    __syncthreads();

    int c = 0;
    for (int w = tid; w < WPR; w += 256) c += __popc(m[w]);
    cnt[tid] = c;
    __syncthreads();
    for (int off = 128; off > 0; off >>= 1) {
        if (tid < off) cnt[tid] += cnt[tid + off];
        __syncthreads();
    }
    int total = cnt[0];
    if (total == 0) {                     // zero-neighbour fallback: sense 0
        if (tid == 0) m[0] = 1u;
        total = 1;
    }
    __syncthreads();
    for (int w = tid; w < WPR; w += 256)
        maskbits[(size_t)row * WPR + w] = m[w];
    if (tid == 0) nsel[row] = total;
}

// ---------------------------------------------------------------- WMMA GEMM
// logits[4096,30000] = hidden_f16[4096,256] @ W_f16[256,30000] + b
// Wt is the K-transposed copy of W: Wt[n*256 + k] -> every fragment half is a
// single K-contiguous global_load_b128.
// One wave computes 3 tiles along N (A reused 3x per K-step, 24 WMMAs/wave);
// 4 waves/block share the same N-tile group so B fragments hit L0 after the
// first wave. The K-loop is software-pipelined with explicit double buffers;
// sched_barrier(0) fences keep the prefetch-load group (step k+1) ahead of the
// WMMA group (step k). NTILES=3 keeps the live set (~24 acc + 48 B + 16 A
// VGPRs) inside the register budget so the pipeline survives RA without
// scratch spills.
__global__ void __launch_bounds__(128, 1)
gemm_wmma_kernel(const _Float16* __restrict__ Ah,
                 const _Float16* __restrict__ Wt,
                 const float* __restrict__ bias,
                 float* __restrict__ out) {
    const int nt0   = blockIdx.x * NTILES;                // 0,3,...,1872
    const int mtile = blockIdx.y * 4 + threadIdx.y;       // 0..255
    const int lane  = threadIdx.x;                        // 0..31
    const int lrow  = lane & 15;
    const int half  = lane >> 4;

    v8f acc[NTILES];
#pragma unroll
    for (int t = 0; t < NTILES; ++t) acc[t] = (v8f){};

    // A fragment (16x32 f16): lanes 0-15 -> K 0..7 & 16..23; lanes 16-31 -> +8
    const _Float16* arow = Ah + (size_t)(mtile * 16 + lrow) * HHDIM + half * 8;
    // B fragment (32x16 f16): lanes 0-15 -> K 0..7 & 16..23; lanes 16-31 -> +8
    const _Float16* brow[NTILES];
#pragma unroll
    for (int t = 0; t < NTILES; ++t)
        brow[t] = Wt + (size_t)((nt0 + t) * 16 + lrow) * HHDIM + half * 8;

    union F { v16h v; v8h h[2]; };
    F a[2], bfrag[2][NTILES];

    // prologue: K-step 0 loads
    a[0].h[0] = *(const v8h*)(arow);
    a[0].h[1] = *(const v8h*)(arow + 16);
#pragma unroll
    for (int t = 0; t < NTILES; ++t) {
        bfrag[0][t].h[0] = *(const v8h*)(brow[t]);
        bfrag[0][t].h[1] = *(const v8h*)(brow[t] + 16);
    }
    __builtin_amdgcn_sched_barrier(0);

#pragma unroll
    for (int ks = 0; ks < HHDIM / 32; ++ks) {             // 8 K-steps
        const int cur = ks & 1, nxt = cur ^ 1;
        if (ks + 1 < HHDIM / 32) {                        // prefetch step ks+1
            const int kk = (ks + 1) * 32;
            a[nxt].h[0] = *(const v8h*)(arow + kk);
            a[nxt].h[1] = *(const v8h*)(arow + kk + 16);
#pragma unroll
            for (int t = 0; t < NTILES; ++t) {
                bfrag[nxt][t].h[0] = *(const v8h*)(brow[t] + kk);
                bfrag[nxt][t].h[1] = *(const v8h*)(brow[t] + kk + 16);
            }
        }
        __builtin_amdgcn_sched_barrier(0);                // loads stay above
#pragma unroll
        for (int t = 0; t < NTILES; ++t) {
            acc[t] = __builtin_amdgcn_wmma_f32_16x16x32_f16(
                false, a[cur].v, false, bfrag[cur][t].v,
                (short)0, acc[t], false, false);
        }
        __builtin_amdgcn_sched_barrier(0);                // MMAs stay above
    }

#pragma unroll
    for (int t = 0; t < NTILES; ++t) {
        const int ncol = (nt0 + t) * 16 + lrow;
        const float bb = bias[ncol];
#pragma unroll
        for (int r = 0; r < 8; ++r) {
            const int mr = mtile * 16 + half * 8 + r;   // C/D layout: VGPR r -> M
            out[(size_t)mr * SSEN + ncol] = acc[t][r] + bb;
        }
    }
}

// ---------------------------------------------------------------- masked log-softmax
// One block per row; three passes over the 120KB row (L2-hot):
//   1) masked max + argmax (lowest index on ties, matching jnp.argmax)
//   2) sum exp(l - max) over mask
//   3) write log-probs in place, delta subtracted at the argmax
__global__ void softmax_finalize_kernel(const unsigned int* __restrict__ maskbits,
                                        const int* __restrict__ nsel,
                                        float* __restrict__ senses) {
    const int row = blockIdx.x;
    const int tid = threadIdx.x;
    float* l = senses + (size_t)row * SSEN;
    const unsigned int* mb = maskbits + (size_t)row * WPR;

    __shared__ float sv[256];
    __shared__ int   si[256];

    // pass 1: masked max / argmax
    float mx = -INFINITY; int ax = 0x7fffffff;
    for (int j = tid; j < SSEN; j += 256) {
        if ((mb[j >> 5] >> (j & 31)) & 1u) {
            float v = l[j];
            if (v > mx || (v == mx && j < ax)) { mx = v; ax = j; }
        }
    }
    sv[tid] = mx; si[tid] = ax;
    __syncthreads();
    for (int off = 128; off > 0; off >>= 1) {
        if (tid < off) {
            float v2 = sv[tid + off]; int i2 = si[tid + off];
            if (v2 > sv[tid] || (v2 == sv[tid] && i2 < si[tid])) {
                sv[tid] = v2; si[tid] = i2;
            }
        }
        __syncthreads();
    }
    mx = sv[0]; ax = si[0];
    __syncthreads();

    // pass 2: masked sum of exp
    float s = 0.f;
    for (int j = tid; j < SSEN; j += 256)
        if ((mb[j >> 5] >> (j & 31)) & 1u) s += expf(l[j] - mx);
    sv[tid] = s;
    __syncthreads();
    for (int off = 128; off > 0; off >>= 1) {
        if (tid < off) sv[tid] += sv[tid + off];
        __syncthreads();
    }
    const float sum   = sv[0];
    const float delta = FEPS * (float)(SSEN - nsel[row]);

    // pass 3: write log-probs in place
    for (int j = tid; j < SSEN; j += 256) {
        float outv;
        if ((mb[j >> 5] >> (j & 31)) & 1u) {
            float p = expf(l[j] - mx) / sum;
            if (j == ax) p -= delta;
            outv = logf(p);
        } else {
            outv = LOGEPS;
        }
        l[j] = outv;
    }
}

// ---------------------------------------------------------------- launcher
extern "C" void kernel_launch(void* const* d_in, const int* in_sizes, int n_in,
                              void* d_out, int out_size, void* d_ws, size_t ws_size,
                              hipStream_t stream) {
    const float* hidden  = (const float*)d_in[0];
    const float* globals = (const float*)d_in[1];
    const float* W       = (const float*)d_in[2];
    const float* b       = (const float*)d_in[3];
    const int*   graph   = (const int*)d_in[4];
    (void)in_sizes; (void)n_in; (void)out_size; (void)ws_size;

    // workspace carve-out (all 256B aligned)
    char* ws = (char*)d_ws;
    size_t off = 0;
    auto take = [&](size_t bytes) {
        void* p = ws + off;
        off += (bytes + 255) & ~(size_t)255;
        return p;
    };
    _Float16*     hid_h = (_Float16*)    take((size_t)NROWS * HHDIM * 2);
    _Float16*     Wt_h  = (_Float16*)    take((size_t)HHDIM * SSEN * 2);
    int*          topk  = (int*)         take((size_t)NROWS * KTOP * 4);
    unsigned int* maskb = (unsigned int*)take((size_t)NROWS * WPR * 4);
    int*          nsel  = (int*)         take((size_t)NROWS * 4);

    float* out_globals = (float*)d_out;
    float* out_senses  = out_globals + (size_t)NROWS * VVOC;

    // 1) hidden f32 -> f16; W f32 -> transposed f16 (K-contiguous per column)
    {
        int n1 = NROWS * HHDIM;
        cvt_f32_f16_kernel<<<(n1 + 255) / 256, 256, 0, stream>>>(hidden, hid_h, n1);
        dim3 tg(SSEN / 16, HHDIM / 16);   // (1875, 16)
        transpose_cvt_kernel<<<tg, dim3(16, 16), 0, stream>>>(W, Wt_h);
    }

    // 2) pass-through predictions_globals (vectorized b128 copies)
    {
        size_t n4 = (size_t)NROWS * VVOC / 4;
        copy_f32x4_kernel<<<(unsigned)((n4 + 255) / 256), 256, 0, stream>>>(
            (const float4*)globals, (float4*)out_globals, n4);
    }

    // 3) top-8 per row
    topk_kernel<<<NROWS, 256, 0, stream>>>(globals, topk);

    // 4) neighbour membership bitmask + counts
    mask_kernel<<<NROWS, 256, 0, stream>>>(graph, topk, maskb, nsel);

    // 5) WMMA GEMM: raw logits staged into the senses half of d_out
    {
        dim3 grid(SSEN / 16 / NTILES /*625*/, NROWS / 16 / 4 /*64*/);
        dim3 block(32, 4);
        gemm_wmma_kernel<<<grid, block, 0, stream>>>(hid_h, Wt_h, b, out_senses);
    }

    // 6) fused masked softmax + delta correction + log, in place per row
    softmax_finalize_kernel<<<NROWS, 256, 0, stream>>>(maskb, nsel, out_senses);
}
